// AttentionHead_33595234189825
// MI455X (gfx1250) — matrix operand order repaired and checked by reading, not compiled, not run on hardware
//
#include <hip/hip_runtime.h>

#define B_ 4
#define S_ 4096
#define D_ 1024
#define H_ 64

// Padded LDS row strides (elements) to spread DS banks; both 16B-aligned.
#define KP_ 72   // K tile row stride (64 + 8)
#define VP_ 40   // V tile row stride (32 + 8)

typedef __attribute__((ext_vector_type(16))) __bf16 v16bf;
typedef __attribute__((ext_vector_type(8)))  __bf16 v8bf;
typedef __attribute__((ext_vector_type(8)))  float  v8f;

static __device__ __forceinline__ v8f wmma_bf16(v16bf a, v16bf b, v8f c) {
  // D = A(16x32 bf16) * B(32x16 bf16) + C(16x16 f32)
  return __builtin_amdgcn_wmma_f32_16x16x32_bf16(false, a, false, b, (short)0, c,
                                                 false, false);
}

// Async DMA of 16 bytes/lane: global -> LDS, tracked by ASYNCcnt.
static __device__ __forceinline__ void async16(unsigned lds_off, const __bf16* g) {
  asm volatile("global_load_async_to_lds_b128 %0, %1, off"
               :: "v"(lds_off), "v"(g) : "memory");
}

// ---------------------------------------------------------------------------
// Kernel 1: QKV projection, fp32 in -> bf16 out.
// grid = (B*S/16, 3), block = 128 (4 waves); wave w owns H columns [16w,16w+16).
// Outputs: q  [B,S,H] bf16, pre-scaled by 1/sqrt(H)=0.125 (folds softmax scale)
//          k  [B,S,H] bf16 (natural layout == scores B-fragment layout)
//          vT [B,H,S] bf16 (transposed    == P@V    B-fragment layout)
// ---------------------------------------------------------------------------
__global__ __launch_bounds__(128)
void qkv_proj_kernel(const float* __restrict__ x,
                     const float* __restrict__ Wq,
                     const float* __restrict__ Wk,
                     const float* __restrict__ Wv,
                     __bf16* __restrict__ qo,
                     __bf16* __restrict__ ko,
                     __bf16* __restrict__ vT) {
  const int lane = threadIdx.x & 31;
  const int wave = threadIdx.x >> 5;       // 0..3 -> H tile
  const int half = lane >> 4;              // 16-lane half
  const int r    = lane & 15;
  const int tile = blockIdx.x;             // row tile over B*S
  const int b    = tile / (S_ / 16);
  const int s0   = (tile % (S_ / 16)) * 16;
  const int h0   = wave * 16;

  const float* W = (blockIdx.y == 0) ? Wq : ((blockIdx.y == 1) ? Wk : Wv);
  const float* xrow = x + ((size_t)b * S_ + s0 + r) * D_;

  v8f acc = {0.f, 0.f, 0.f, 0.f, 0.f, 0.f, 0.f, 0.f};

  for (int c = 0; c < D_; c += 32) {
    v16bf a, bm;
#pragma unroll
    for (int i = 0; i < 8; ++i) {
      // A fragment: lane = row; K = 16*half + s (16 consecutive K/lane)
      a[i]     = (__bf16)xrow[c + 16 * half + i];
      a[8 + i] = (__bf16)xrow[c + 16 * half + 8 + i];
    }
#pragma unroll
    for (int e = 0; e < 16; ++e) {
      // B fragment: lane = column r; K = 16*half + e (16 consecutive K/lane)
      bm[e] = (__bf16)W[(size_t)(c + 16 * half + e) * H_ + h0 + r];
    }
    acc = wmma_bf16(a, bm, acc);
  }

  if (blockIdx.y == 2) {
    // V transposed: [B, H, S]
#pragma unroll
    for (int j = 0; j < 8; ++j) {
      const int row = j + 8 * half;   // S offset within tile
      vT[((size_t)b * H_ + h0 + r) * S_ + s0 + row] = (__bf16)acc[j];
    }
  } else {
    const float scale = (blockIdx.y == 0) ? 0.125f : 1.0f;
    __bf16* dst = (blockIdx.y == 0) ? qo : ko;
#pragma unroll
    for (int j = 0; j < 8; ++j) {
      const int row = j + 8 * half;
      dst[((size_t)b * S_ + s0 + row) * H_ + h0 + r] = (__bf16)(acc[j] * scale);
    }
  }
}

// ---------------------------------------------------------------------------
// Kernel 2: cooperative causal flash attention.
// grid = B*S/64, block = 128 (4 waves). The block owns 64 queries (wave w ->
// rows Q0+16w..Q0+16w+15), so the causal trip count is workgroup-uniform and
// the 32-key K/V tiles are SHARED: staged into double-buffered LDS with
// global_load_async_to_lds_b128 (ASYNCcnt), retired with s_wait_asynccnt +
// barrier, overlapping the DMA of block i+1 with the 8 WMMAs of block i.
// ---------------------------------------------------------------------------
__global__ __launch_bounds__(128)
void attn_kernel(const __bf16* __restrict__ q,
                 const __bf16* __restrict__ k,
                 const __bf16* __restrict__ vT,
                 float* __restrict__ out) {
  __shared__ __align__(16) __bf16 kt[2][32 * KP_];   // 32 keys x 64 h (padded)
  __shared__ __align__(16) __bf16 vt[2][64 * VP_];   // 64 h x 32 keys (padded)
  __shared__ __align__(16) __bf16 pbuf[4][16 * 32];  // per-wave P transpose

  const int lane = threadIdx.x & 31;
  const int wave = threadIdx.x >> 5;
  const int half = lane >> 4;
  const int r    = lane & 15;
  const int bq   = blockIdx.x;                 // 64-query block over B*S
  const int b    = bq / (S_ / 64);
  const int Q0   = (bq % (S_ / 64)) * 64;
  const int q0   = Q0 + wave * 16;             // this wave's 16 query rows
  const int kend = Q0 + 64;                    // workgroup-uniform causal bound

  const __bf16* kb  = k  + (size_t)b * S_ * H_;
  const __bf16* vTb = vT + (size_t)b * H_ * S_;

  // ---- async stage of one 32-key block into LDS buffer bsel ----
  auto stage = [&](int bsel, int k0) {
    const unsigned kbase = (unsigned)(uintptr_t)&kt[bsel][0];
    const unsigned vbase = (unsigned)(uintptr_t)&vt[bsel][0];
    // K tile: 32 rows x 128B, contiguous 4KB in global; 256 x 16B chunks.
#pragma unroll
    for (int u = 0; u < 2; ++u) {
      const int c   = wave * 64 + lane + u * 32;
      const int row = c >> 3, col = c & 7;
      async16(kbase + (unsigned)(row * KP_ + col * 8) * 2,
              kb + (size_t)k0 * H_ + c * 8);
    }
    // V tile: 64 rows x 64B, row stride S_ in global; 256 x 16B chunks.
#pragma unroll
    for (int u = 0; u < 2; ++u) {
      const int c   = wave * 64 + lane + u * 32;
      const int row = c >> 2, col = c & 3;
      async16(vbase + (unsigned)(row * VP_ + col * 8) * 2,
              vTb + (size_t)row * S_ + k0 + col * 8);
    }
  };

  // Q A-fragments (already pre-scaled by 0.125 at projection time)
  const __bf16* qrow = q + ((size_t)b * S_ + q0 + r) * H_;
  v16bf aq[2];
#pragma unroll
  for (int c = 0; c < 2; ++c) {
    const v8bf lo = *(const v8bf*)(qrow + 32 * c + 16 * half);
    const v8bf hi = *(const v8bf*)(qrow + 32 * c + 16 * half + 8);
#pragma unroll
    for (int i = 0; i < 8; ++i) { aq[c][i] = lo[i]; aq[c][8 + i] = hi[i]; }
  }

  v8f o[4];
#pragma unroll
  for (int t = 0; t < 4; ++t) o[t] = (v8f){0.f,0.f,0.f,0.f,0.f,0.f,0.f,0.f};
  float m[8], l[8];
#pragma unroll
  for (int j = 0; j < 8; ++j) { m[j] = -3.0e38f; l[j] = 0.f; }

  __bf16* pwb = pbuf[wave];

  stage(0, 0);                                 // prologue DMA for block 0
  int buf = 0;

  for (int k0 = 0; k0 < kend; k0 += 32) {
    // retire this block's DMA (issued last iteration / prologue), then make
    // the tiles visible to all 4 waves.
    asm volatile("s_wait_asynccnt 0" ::: "memory");
    __syncthreads();
    // overlap: issue next block's DMA into the other buffer
    if (k0 + 32 < kend) stage(buf ^ 1, k0 + 32);

    const __bf16* ktb = kt[buf];
    const __bf16* vtb = vt[buf];

    // ---- scores: two 16-key subtiles, K = 64 over H (2 WMMAs each) ----
    v8f s[2];
#pragma unroll
    for (int n = 0; n < 2; ++n) {
      const int kl  = 16 * n + r;              // local key (B-frag column)
      const int key = k0 + kl;                 // global key index
      const v16bf bk0 = *(const v16bf*)(ktb + kl * KP_ + 16 * half);
      const v16bf bk1 = *(const v16bf*)(ktb + kl * KP_ + 32 + 16 * half);
      v8f sc = {0.f, 0.f, 0.f, 0.f, 0.f, 0.f, 0.f, 0.f};
      sc = wmma_bf16(aq[0], bk0, sc);
      sc = wmma_bf16(aq[1], bk1, sc);
      // causal mask (select after MMA; EXEC all-ones at every WMMA)
#pragma unroll
      for (int j = 0; j < 8; ++j) {
        const int row_g = q0 + j + 8 * half;
        sc[j] = (key > row_g) ? -1.0e30f : sc[j];
      }
      s[n] = sc;
    }

    // ---- online softmax: rows live across 16-lane halves ----
    float alpha[8];
#pragma unroll
    for (int j = 0; j < 8; ++j) {
      float bmax = fmaxf(s[0][j], s[1][j]);
#pragma unroll
      for (int off = 1; off < 16; off <<= 1)
        bmax = fmaxf(bmax, __shfl_xor(bmax, off, 32));
      const float mn = fmaxf(m[j], bmax);
      alpha[j] = __expf(m[j] - mn);
      const float p0 = __expf(s[0][j] - mn);
      const float p1 = __expf(s[1][j] - mn);
      float rs = p0 + p1;
#pragma unroll
      for (int off = 1; off < 16; off <<= 1)
        rs += __shfl_xor(rs, off, 32);
      l[j] = l[j] * alpha[j] + rs;
      m[j] = mn;
      s[0][j] = p0;
      s[1][j] = p1;
#pragma unroll
      for (int t = 0; t < 4; ++t) o[t][j] *= alpha[j];
    }

    // ---- transpose P: C layout -> bf16 row-major [16][32] in LDS ----
#pragma unroll
    for (int j = 0; j < 8; ++j) {
      const int row = j + 8 * half;
      pwb[row * 32 + r]      = (__bf16)s[0][j];
      pwb[row * 32 + 16 + r] = (__bf16)s[1][j];
    }
    asm volatile("s_wait_dscnt 0" ::: "memory");   // wave-local ordering
    v16bf ap;
    {
      const v8bf lo = *(const v8bf*)(pwb + r * 32 + 16 * half);
      const v8bf hi = *(const v8bf*)(pwb + r * 32 + 16 * half + 8);
#pragma unroll
      for (int i = 0; i < 8; ++i) { ap[i] = lo[i]; ap[8 + i] = hi[i]; }
    }

    // ---- O += P @ V  (4 H tiles of 16) ----
#pragma unroll
    for (int t = 0; t < 4; ++t) {
      const v16bf bv = *(const v16bf*)(vtb + (16 * t + r) * VP_ + 16 * half);
      o[t] = wmma_bf16(ap, bv, o[t]);
    }

    buf ^= 1;
  }

  // ---- normalize and store (fp32 output) ----
#pragma unroll
  for (int j = 0; j < 8; ++j) {
    const float inv = 1.0f / l[j];
    const int row_g = q0 + j + 8 * half;
    float* orow = out + ((size_t)b * S_ + row_g) * H_;
#pragma unroll
    for (int t = 0; t < 4; ++t) orow[16 * t + r] = o[t][j] * inv;
  }
}

// ---------------------------------------------------------------------------
extern "C" void kernel_launch(void* const* d_in, const int* in_sizes, int n_in,
                              void* d_out, int out_size, void* d_ws, size_t ws_size,
                              hipStream_t stream) {
  (void)in_sizes; (void)n_in; (void)out_size; (void)ws_size;
  const float* x  = (const float*)d_in[0];
  const float* Wq = (const float*)d_in[1];
  const float* Wk = (const float*)d_in[2];
  const float* Wv = (const float*)d_in[3];
  float* out = (float*)d_out;

  const size_t proj_elems = (size_t)B_ * S_ * H_;   // 1 Mi elements each
  __bf16* qb = (__bf16*)d_ws;
  __bf16* kb = qb + proj_elems;
  __bf16* vb = kb + proj_elems;

  hipLaunchKernelGGL(qkv_proj_kernel, dim3(B_ * S_ / 16, 3), dim3(128), 0, stream,
                     x, Wq, Wk, Wv, qb, kb, vb);
  hipLaunchKernelGGL(attn_kernel, dim3(B_ * S_ / 64), dim3(128), 0, stream,
                     qb, kb, vb, out);
}
